// SGNS_60722247631361
// MI455X (gfx1250) — matrix operand order repaired and checked by reading, not compiled, not run on hardware
//
#include <hip/hip_runtime.h>
#include <hip/hip_bf16.h>
#include <math.h>

typedef __attribute__((ext_vector_type(16))) _Float16 v16h;
typedef __attribute__((ext_vector_type(8)))  float    v8f;

#define EMB_D 128

// Stable log(sigmoid(x)) = min(x,0) - log1p(exp(-|x|))
__device__ __forceinline__ float log_sigmoid(float x) {
    return fminf(x, 0.0f) - log1pf(__expf(-fabsf(x)));
}

// One wave32 per batch row. Broadcast-A WMMA trick:
//   A[m][k] = ivec[k] for all m  (contiguous K order, same as B)
//   B[k][n] = ctx_n[k]           (16 gathered context vectors as columns)
//   D[m][n] = dot(ivec, ctx_n)   -> lane L reads acc[0] = dot for ctx (L%16)
__global__ void sgns_wmma_kernel(const float* __restrict__ emb_i,
                                 const float* __restrict__ emb_o,
                                 const int*   __restrict__ iword,
                                 const int*   __restrict__ owords,
                                 const int*   __restrict__ nwords,
                                 float*       __restrict__ row_loss,
                                 int B, int C, int NCK) {
    const int lane = threadIdx.x & 31;
    const int wv   = threadIdx.x >> 5;
    const int b    = blockIdx.x * 8 + wv;
    if (b >= B) return;                       // wave-uniform exit

    const int iw = iword[b];
    const float* __restrict__ irow = emb_i + (size_t)iw * EMB_D;

    // ---- Build broadcast-A chunks (f16), in contiguous K order (same as B):
    // lanes 0-15:  VGPR0-3 -> K = c*32 + 0..7  ; VGPR4-7 -> K = c*32 + 8..15
    // lanes 16-31: VGPR0-3 -> K = c*32 + 16..23; VGPR4-7 -> K = c*32 + 24..31
    const int h = (lane & 16) ? 16 : 0;
    v16h A[4];
#pragma unroll
    for (int c = 0; c < 4; ++c) {
        const float4* p0 = reinterpret_cast<const float4*>(irow + c * 32 + h);
        const float4* p1 = reinterpret_cast<const float4*>(irow + c * 32 + 8 + h);
        float4 x0 = p0[0], x1 = p0[1];
        float4 y0 = p1[0], y1 = p1[1];
        v16h a;
        a[0]  = (_Float16)x0.x; a[1]  = (_Float16)x0.y;
        a[2]  = (_Float16)x0.z; a[3]  = (_Float16)x0.w;
        a[4]  = (_Float16)x1.x; a[5]  = (_Float16)x1.y;
        a[6]  = (_Float16)x1.z; a[7]  = (_Float16)x1.w;
        a[8]  = (_Float16)y0.x; a[9]  = (_Float16)y0.y;
        a[10] = (_Float16)y0.z; a[11] = (_Float16)y0.w;
        a[12] = (_Float16)y1.x; a[13] = (_Float16)y1.y;
        a[14] = (_Float16)y1.z; a[15] = (_Float16)y1.w;
        A[c] = a;
    }

    const int total   = C + NCK;              // 210
    const int ngroups = (total + 15) >> 4;    // 14
    const int nsub    = lane & 15;            // this lane's context column
    const int koff    = (lane & 16) ? 16 : 0; // B-matrix half-wave K offset

    float accsum = 0.0f;

    for (int g = 0; g < ngroups; ++g) {
        const int j = (g << 4) + nsub;        // combined index: [owords | nwords]
        int w = 0;
        if (j < C)          w = owords[b * C + j];
        else if (j < total) w = nwords[b * NCK + (j - C)];
        const float* __restrict__ crow = emb_o + (size_t)w * EMB_D;

        v8f acc = {};                         // C-matrix starts at 0
#pragma unroll
        for (int c = 0; c < 4; ++c) {
            // B layout: lanes 0-15 hold K = c*32 + 0..15 of column (lane),
            //           lanes 16-31 hold K = c*32 + 16..31 of column (lane-16)
            const float4* p = reinterpret_cast<const float4*>(crow + c * 32 + koff);
            float4 x0 = p[0], x1 = p[1], x2 = p[2], x3 = p[3];
            v16h bm;
            bm[0]  = (_Float16)x0.x; bm[1]  = (_Float16)x0.y;
            bm[2]  = (_Float16)x0.z; bm[3]  = (_Float16)x0.w;
            bm[4]  = (_Float16)x1.x; bm[5]  = (_Float16)x1.y;
            bm[6]  = (_Float16)x1.z; bm[7]  = (_Float16)x1.w;
            bm[8]  = (_Float16)x2.x; bm[9]  = (_Float16)x2.y;
            bm[10] = (_Float16)x2.z; bm[11] = (_Float16)x2.w;
            bm[12] = (_Float16)x3.x; bm[13] = (_Float16)x3.y;
            bm[14] = (_Float16)x3.z; bm[15] = (_Float16)x3.w;
            acc = __builtin_amdgcn_wmma_f32_16x16x32_f16(
                /*neg_a=*/false, A[c], /*neg_b=*/false, bm,
                /*c_mod=*/(short)0, acc, /*reuse_a=*/false, /*reuse_b=*/false);
        }

        // D[0][nsub] lives in acc element 0 of every lane (rows identical).
        const float d = acc[0];
        float t = 0.0f;
        if (lane < 16 && j < total) {         // lanes 16-31 are duplicates
            const float x = (j < C) ? d : -d; // negatives: log_sigmoid(-dot)
            t = log_sigmoid(x);
        }
        accsum += t;
    }

    // Wave32 reduction (fixed order -> deterministic)
#pragma unroll
    for (int m = 16; m >= 1; m >>= 1)
        accsum += __shfl_xor(accsum, m, 32);

    if (lane == 0)
        row_loss[b] = -accsum / (float)C;     // -(oloss + nloss) for this row
}

// Deterministic final reduction: single block, fixed-order strided sums + tree.
__global__ void sgns_reduce_kernel(const float* __restrict__ row_loss,
                                   float* __restrict__ out, int n) {
    __shared__ float sm[256];
    const int t = threadIdx.x;
    float s = 0.0f;
    for (int i = t; i < n; i += 256) s += row_loss[i];
    sm[t] = s;
    __syncthreads();
    for (int k = 128; k > 0; k >>= 1) {
        if (t < k) sm[t] += sm[t + k];
        __syncthreads();
    }
    if (t == 0) out[0] = sm[0] / (float)n;    // mean over batch
}

extern "C" void kernel_launch(void* const* d_in, const int* in_sizes, int n_in,
                              void* d_out, int out_size, void* d_ws, size_t ws_size,
                              hipStream_t stream) {
    const float* emb_i  = (const float*)d_in[0];
    const float* emb_o  = (const float*)d_in[1];
    const int*   iword  = (const int*)d_in[2];
    const int*   owords = (const int*)d_in[3];
    const int*   nwords = (const int*)d_in[4];

    const int B   = in_sizes[2];
    const int C   = in_sizes[3] / B;          // 10
    const int NCK = in_sizes[4] / B;          // C*K = 200

    float* row_loss = (float*)d_ws;           // B floats of scratch
    float* out      = (float*)d_out;

    const int blocks = (B + 7) / 8;           // 8 waves (rows) per 256-thread block
    sgns_wmma_kernel<<<blocks, 256, 0, stream>>>(
        emb_i, emb_o, iword, owords, nwords, row_loss, B, C, NCK);

    sgns_reduce_kernel<<<1, 256, 0, stream>>>(row_loss, out, B);
}
